// LightInvariantPointAttention_23905787969872
// MI455X (gfx1250) — matrix operand order repaired and checked by reading, not compiled, not run on hardware
//
#include <hip/hip_runtime.h>
#include <hip/hip_bf16.h>

typedef __attribute__((ext_vector_type(16))) _Float16 v16h;
typedef __attribute__((ext_vector_type(8)))  _Float16 h8;
typedef __attribute__((ext_vector_type(8)))  float    v8f;

#define B_ 4
#define N_ 512
#define DIM_ 384
#define H_ 8
#define DK_ 16
#define DV_ 16
#define PK_ 4
#define PV_ 8
#define PD_ 64
#define TSTR 40   // padded LDS stride (halves) for transposed tiles: 80B, 16B-aligned

// SCALAR_SCALE = (3*16)^-0.5, POINT_SCALE = (3*4*4.5)^-0.5, PAIR_SCALE = 3^-0.5
#define SCALAR_SCALE_ 0.14433756729740643f
#define POINT_SCALE_  0.13608276348795434f
#define PAIR_SCALE_   0.5773502691896258f
#define NEG_MAX_ (-3.402823466e38f)

__device__ __forceinline__ v8f wmma16(v16h a, v16h b, v8f c) {
  return __builtin_amdgcn_wmma_f32_16x16x32_f16(false, a, false, b, (short)0, c,
                                                false, false);
}

// load 16 contiguous halves (two b128s) into a WMMA operand register set
__device__ __forceinline__ v16h ld16h(const _Float16* p) {
  h8 lo = *(const h8*)p;
  h8 hi = *(const h8*)(p + 8);
  v16h v;
#pragma unroll
  for (int e = 0; e < 8; ++e) { v[e] = lo[e]; v[8 + e] = hi[e]; }
  return v;
}

// ---------------------------------------------------------------- f32 -> f16
__global__ void k_f2h(const float* __restrict__ s, _Float16* __restrict__ d, int n) {
  int i = blockIdx.x * blockDim.x + threadIdx.x;
  if (i < n) d[i] = (_Float16)s[i];
}

// ------------------------------------------------- generic WMMA GEMM (f16->f32)
// C[M x Nc] = A[M x K] * Bm[K x Nc] (+ bias[col]).  One wave per 16x16 C tile.
// B is staged TRANSPOSED into LDS (coalesced b128 global loads, lane-contiguous
// b16 scatters), so the B operand is two ds_load_b128 per WMMA.
__global__ void __launch_bounds__(32)
k_gemm_f16(const _Float16* __restrict__ A, const _Float16* __restrict__ Bm,
           float* __restrict__ C, const float* __restrict__ bias,
           int M, int Nc, int K) {
  (void)M;
  __shared__ __align__(16) _Float16 Btt[16 * TSTR];   // [n][k] transposed
  int ntn = Nc >> 4;
  int tm = blockIdx.x / ntn, tn = blockIdx.x % ntn;
  int l = threadIdx.x & 31, r = l & 15, hs = l >> 4;
  int col = tn * 16 + r;
  v8f acc = {};
  for (int k0 = 0; k0 < K; k0 += 32) {
    const _Float16* pbs = Bm + (size_t)(k0 + l) * Nc + tn * 16;
    h8 blo = *(const h8*)pbs;
    h8 bhi = *(const h8*)(pbs + 8);
    __syncthreads();
#pragma unroll
    for (int e = 0; e < 8; ++e) {
      Btt[e * TSTR + l]       = blo[e];
      Btt[(8 + e) * TSTR + l] = bhi[e];
    }
    __syncthreads();
    // A tile: lane (r,hs) holds row r, K = k0+hs*16..+15
    const _Float16* pa = A + (size_t)(tm * 16 + r) * K + k0 + hs * 16;
    h8 lo = *(const h8*)pa;
    h8 hi = *(const h8*)(pa + 8);
    v16h a;
#pragma unroll
    for (int e = 0; e < 8; ++e) { a[e] = lo[e]; a[8 + e] = hi[e]; }
    v16h b = ld16h(&Btt[r * TSTR + hs * 16]);
    acc = wmma16(a, b, acc);
  }
  float bv = bias ? bias[col] : 0.0f;
#pragma unroll
  for (int v = 0; v < 8; ++v) {
    int row = tm * 16 + v + 8 * hs;
    C[(size_t)row * Nc + col] = acc[v] + bv;
  }
}

// ------------------------------------------------------------- geometry / RoPE
__global__ void k_geom(const float* __restrict__ Pqs, const float* __restrict__ Pks,
                       const float* __restrict__ Pvs, const float* __restrict__ Pqp,
                       const float* __restrict__ Pkp, const float* __restrict__ Pvp,
                       const float* __restrict__ rot, const float* __restrict__ trans,
                       const int* __restrict__ pos_ids, const float* __restrict__ point_w,
                       _Float16* __restrict__ qA, _Float16* __restrict__ kB,
                       _Float16* __restrict__ vsh, _Float16* __restrict__ vph,
                       float* __restrict__ c1, float* __restrict__ c2) {
  int t = blockIdx.x * blockDim.x + threadIdx.x;
  if (t >= B_ * N_ * H_) return;
  int h = t & 7; int bn = t >> 3; int n = bn & (N_ - 1); int b = bn >> 9;
  size_t row = (size_t)bn;                     // b*N + n
  size_t bhn = ((size_t)b * H_ + h) * N_ + n;  // (b,h,n)
  int pos = pos_ids[row];
  const float* pq = Pqs + row * (H_ * DK_) + h * DK_;
  const float* pk = Pks + row * (H_ * DK_) + h * DK_;
  float qr[16], kr[16];
#pragma unroll
  for (int d = 0; d < 8; ++d) {
    float fr = __expf(-((float)(2 * d) / 16.0f) * 9.210340371976184f); // ln 1e4
    float ss, cc;
    __sincosf((float)pos * fr, &ss, &cc);
    qr[d]     = pq[d] * cc     - pq[d + 8] * ss;
    qr[d + 8] = pq[d + 8] * cc + pq[d] * ss;
    kr[d]     = pk[d] * cc     - pk[d + 8] * ss;
    kr[d + 8] = pk[d + 8] * cc + pk[d] * ss;
  }
  float pw = log1pf(__expf(point_w[h]));   // softplus
  float sA = pw * POINT_SCALE_;
  const float* R = rot + row * 9;          // rot[b,n,c,r] -> R[c*3+r]
  const float* T = trans + row * 3;
  float qq = 0.f, kk = 0.f;
  float qp[12], kp[12];
#pragma unroll
  for (int d = 0; d < PK_; ++d) {
    const float* aq = Pqp + row * (H_ * PK_ * 3) + h * (PK_ * 3) + d * 3;
    const float* ak = Pkp + row * (H_ * PK_ * 3) + h * (PK_ * 3) + d * 3;
#pragma unroll
    for (int rr = 0; rr < 3; ++rr) {
      float vq = aq[0] * R[0 * 3 + rr] + aq[1] * R[1 * 3 + rr] + aq[2] * R[2 * 3 + rr] + T[rr];
      float vk = ak[0] * R[0 * 3 + rr] + ak[1] * R[1 * 3 + rr] + ak[2] * R[2 * 3 + rr] + T[rr];
      qp[d * 3 + rr] = vq; kp[d * 3 + rr] = vk;
      qq += vq * vq; kk += vk * vk;
    }
  }
  _Float16* qrow = qA + bhn * 32;
  _Float16* krow = kB + bhn * 32;
#pragma unroll
  for (int d = 0; d < 16; ++d) {
    qrow[d] = (_Float16)(qr[d] * SCALAR_SCALE_);
    krow[d] = (_Float16)kr[d];
  }
#pragma unroll
  for (int e = 0; e < 12; ++e) {
    qrow[16 + e] = (_Float16)(qp[e] * sA);
    krow[16 + e] = (_Float16)kp[e];
  }
#pragma unroll
  for (int e = 0; e < 4; ++e) { qrow[28 + e] = (_Float16)0.f; krow[28 + e] = (_Float16)0.f; }
  c1[bhn] = -0.5f * sA * qq;
  c2[bhn] = -0.5f * sA * kk;
  const float* pv = Pvs + row * (H_ * DV_) + h * DV_;
#pragma unroll
  for (int d = 0; d < 16; ++d) vsh[bhn * 16 + d] = (_Float16)pv[d];
#pragma unroll
  for (int d = 0; d < PV_; ++d) {
    const float* av = Pvp + row * (H_ * PV_ * 3) + h * (PV_ * 3) + d * 3;
#pragma unroll
    for (int rr = 0; rr < 3; ++rr) {
      float v = av[0] * R[0 * 3 + rr] + av[1] * R[1 * 3 + rr] + av[2] * R[2 * 3 + rr] + T[rr];
      vph[bhn * 24 + d * 3 + rr] = (_Float16)v;
    }
  }
}

// -------------------------------------------------- pair bias (single 268MB pass)
__global__ void k_pairbias(const float* __restrict__ pairwise,
                           const float* __restrict__ W_pair,
                           const float* __restrict__ b_pair,
                           _Float16* __restrict__ bias8) {
  __shared__ float Wp[PD_ * H_];
  __shared__ float bp[H_];
  int tid = threadIdx.x;
  for (int i = tid; i < PD_ * H_; i += blockDim.x) Wp[i] = W_pair[i];
  if (tid < H_) bp[tid] = b_pair[tid];
  __syncthreads();
  int t = blockIdx.x * blockDim.x + tid;
  int h = t & 7;
  size_t rowv = (size_t)(t >> 3);
  const float4* pr4 = (const float4*)(pairwise + rowv * PD_);
  float s = bp[h];
#pragma unroll
  for (int c = 0; c < PD_ / 4; ++c) {
    float4 v = pr4[c];
    s += v.x * Wp[(c * 4 + 0) * H_ + h];
    s += v.y * Wp[(c * 4 + 1) * H_ + h];
    s += v.z * Wp[(c * 4 + 2) * H_ + h];
    s += v.w * Wp[(c * 4 + 3) * H_ + h];
  }
  bias8[(size_t)t] = (_Float16)(s * PAIR_SCALE_);
}

// ------------------------------------------------------------------- attention
// One wave per (b, h, 16-row i-tile). Fused scalar+point logits in one WMMA,
// LDS softmax (f32), f16 prob tile in LDS for WMMA-A, transposed V tiles for
// WMMA-B: every operand is loaded as b128 pairs.
__global__ void __launch_bounds__(32)
k_attn(const _Float16* __restrict__ qA, const _Float16* __restrict__ kB,
       const _Float16* __restrict__ bias8, const float* __restrict__ c1,
       const float* __restrict__ c2, const _Float16* __restrict__ vsh,
       const _Float16* __restrict__ vph, const unsigned char* __restrict__ mask,
       _Float16* __restrict__ attn, float* __restrict__ rs, float* __restrict__ rpts) {
  __shared__ float P[16][N_ + 8];                    // f32 logits / probs
  __shared__ __align__(16) _Float16 Pa[16][N_ + 8];  // f16 probs (WMMA A)
  __shared__ __align__(16) _Float16 Vst[16 * TSTR];  // v_s transposed [d][j]
  __shared__ __align__(16) _Float16 Vpt[32 * TSTR];  // v_p transposed [e][j], rows 24..31 zero
  int it = blockIdx.x & 31;
  int h  = (blockIdx.x >> 5) & 7;
  int b  = blockIdx.x >> 8;
  int l = threadIdx.x & 31, r = l & 15, hs = l >> 4;
  int i0 = it * 16;
  size_t bhN = ((size_t)b * H_ + h) * N_;
  // zero the v_p padding rows once
  for (int idx = l; idx < 8 * TSTR; idx += 32) Vpt[24 * TSTR + idx] = (_Float16)0.f;
  v16h a;
  {
    const _Float16* pa = qA + (bhN + i0 + r) * 32 + hs * 16;
    h8 lo = *(const h8*)pa;
    h8 hi = *(const h8*)(pa + 8);
#pragma unroll
    for (int e = 0; e < 8; ++e) { a[e] = lo[e]; a[8 + e] = hi[e]; }
  }
  float c1v[8]; float badrow[8];
#pragma unroll
  for (int v = 0; v < 8; ++v) {
    int i = i0 + v + 8 * hs;
    c1v[v] = c1[bhN + i];
    badrow[v] = mask[(size_t)b * N_ + i] ? 0.f : 1.f;
  }
  for (int jt = 0; jt < N_ / 16; ++jt) {
    int j = jt * 16 + r;
    v16h bvec = ld16h(kB + (bhN + j) * 32 + hs * 16);
    v8f z = {};
    v8f s = wmma16(a, bvec, z);
    float c2j = c2[bhN + j];
    bool mj = mask[(size_t)b * N_ + j] != 0;
#pragma unroll
    for (int v = 0; v < 8; ++v) {
      int i = i0 + v + 8 * hs;
      float lg = s[v] + c1v[v] + c2j
               + (float)bias8[(((size_t)b * N_ + i) * N_ + j) * 8 + h];
      if (badrow[v] != 0.f || !mj) lg = NEG_MAX_;
      P[v + 8 * hs][j] = lg;
    }
  }
  __syncthreads();
  for (int rr = 0; rr < 16; ++rr) {
    float m = NEG_MAX_;
    for (int c = l; c < N_; c += 32) m = fmaxf(m, P[rr][c]);
#pragma unroll
    for (int off = 16; off > 0; off >>= 1) m = fmaxf(m, __shfl_xor(m, off, 32));
    float sum = 0.f;
    for (int c = l; c < N_; c += 32) { float e = __expf(P[rr][c] - m); P[rr][c] = e; sum += e; }
#pragma unroll
    for (int off = 16; off > 0; off >>= 1) sum += __shfl_xor(sum, off, 32);
    float inv = 1.0f / sum;
    for (int c = l; c < N_; c += 32) {
      float p = P[rr][c] * inv;
      Pa[rr][c] = (_Float16)p;
      attn[(bhN + i0 + rr) * N_ + c] = (_Float16)p;
    }
  }
  __syncthreads();
  v8f cs = {}, cp0 = {}, cp1 = {};
  for (int kt = 0; kt < N_ / 32; ++kt) {
    int k0 = kt * 32;
    // stage v_s / v_p rows j = k0+l, transposed, into LDS
    const _Float16* sv = vsh + (bhN + k0 + l) * 16;
    h8 s0 = *(const h8*)sv;
    h8 s1 = *(const h8*)(sv + 8);
    const _Float16* sp = vph + (bhN + k0 + l) * 24;
    h8 p0 = *(const h8*)sp;
    h8 p1 = *(const h8*)(sp + 8);
    h8 p2 = *(const h8*)(sp + 16);
    __syncthreads();
#pragma unroll
    for (int e = 0; e < 8; ++e) {
      Vst[e * TSTR + l]        = s0[e];
      Vst[(8 + e) * TSTR + l]  = s1[e];
      Vpt[e * TSTR + l]        = p0[e];
      Vpt[(8 + e) * TSTR + l]  = p1[e];
      Vpt[(16 + e) * TSTR + l] = p2[e];
    }
    __syncthreads();
    v16h aa  = ld16h(&Pa[r][k0 + hs * 16]);
    v16h bs  = ld16h(&Vst[r * TSTR + hs * 16]);
    v16h bp0 = ld16h(&Vpt[r * TSTR + hs * 16]);
    v16h bp1 = ld16h(&Vpt[(16 + r) * TSTR + hs * 16]);   // rows >=24 are zeros
    cs  = wmma16(aa, bs, cs);
    cp0 = wmma16(aa, bp0, cp0);
    cp1 = wmma16(aa, bp1, cp1);
  }
#pragma unroll
  for (int v = 0; v < 8; ++v) {
    int i = i0 + v + 8 * hs;
    rs[(bhN + i) * 16 + r] = cs[v];
    rpts[(bhN + i) * 24 + r] = cp0[v];
    if (r < 8) rpts[(bhN + i) * 24 + 16 + r] = cp1[v];
  }
}

// ------------------------------------------- res_pair = attn @ pairwise[b,i,:,:]
// One wave per (b,i); heads on the M axis (8 valid rows), 4 N-tiles of d.
// The 268MB pairwise stream is loaded coalesced (b128), converted to f16, and
// staged TRANSPOSED so each WMMA B operand is two ds_load_b128.
__global__ void __launch_bounds__(32)
k_respair(const _Float16* __restrict__ attn, const float* __restrict__ pairwise,
          float* __restrict__ rpair) {
  __shared__ __align__(16) _Float16 Ptt[PD_ * TSTR];   // [d][j] transposed
  int b = blockIdx.x >> 9;
  int i = blockIdx.x & (N_ - 1);
  int l = threadIdx.x & 31, r = l & 15, hs = l >> 4;
  v8f c0 = {}, c1v = {}, c2v = {}, c3v = {};
  const float* prow = pairwise + (((size_t)b * N_ + i) * N_) * PD_;
  for (int kt = 0; kt < N_ / 32; ++kt) {
    int k0 = kt * 32;
    const float4* src = (const float4*)(prow + (size_t)(k0 + l) * PD_);
    if (kt + 1 < N_ / 32) {   // prefetch next tile's rows (gfx1250 global_prefetch)
      const float* nxt = prow + (size_t)(k0 + 32 + l) * PD_;
      __builtin_prefetch(nxt, 0, 1);
      __builtin_prefetch(nxt + 32, 0, 1);
    }
    __syncthreads();
#pragma unroll
    for (int c4 = 0; c4 < PD_ / 4; ++c4) {
      float4 v = src[c4];
      Ptt[(c4 * 4 + 0) * TSTR + l] = (_Float16)v.x;
      Ptt[(c4 * 4 + 1) * TSTR + l] = (_Float16)v.y;
      Ptt[(c4 * 4 + 2) * TSTR + l] = (_Float16)v.z;
      Ptt[(c4 * 4 + 3) * TSTR + l] = (_Float16)v.w;
    }
    __syncthreads();
    v16h aa;
    if (r < H_) {
      const _Float16* pa = attn + (((size_t)b * H_ + r) * N_ + i) * N_ + k0 + hs * 16;
      h8 lo = *(const h8*)pa;
      h8 hi = *(const h8*)(pa + 8);
#pragma unroll
      for (int e = 0; e < 8; ++e) { aa[e] = lo[e]; aa[8 + e] = hi[e]; }
    } else {
#pragma unroll
      for (int e = 0; e < 16; ++e) aa[e] = (_Float16)0.f;
    }
    v16h b0 = ld16h(&Ptt[r * TSTR + hs * 16]);
    v16h b1 = ld16h(&Ptt[(16 + r) * TSTR + hs * 16]);
    v16h b2 = ld16h(&Ptt[(32 + r) * TSTR + hs * 16]);
    v16h b3 = ld16h(&Ptt[(48 + r) * TSTR + hs * 16]);
    c0  = wmma16(aa, b0, c0);
    c1v = wmma16(aa, b1, c1v);
    c2v = wmma16(aa, b2, c2v);
    c3v = wmma16(aa, b3, c3v);
  }
  if (hs == 0) {   // lanes 0..15: C rows M=v are heads 0..7
#pragma unroll
    for (int v = 0; v < 8; ++v) {
      float* o = rpair + (((size_t)b * H_ + v) * N_ + i) * PD_;
      o[r] = c0[v]; o[16 + r] = c1v[v]; o[32 + r] = c2v[v]; o[48 + r] = c3v[v];
    }
  }
}

// --------------------------------------- inverse rotation, norms, feature concat
__global__ void k_finalize(const float* __restrict__ rs, const float* __restrict__ rpts,
                           const float* __restrict__ rpair, const float* __restrict__ rot,
                           const float* __restrict__ trans, _Float16* __restrict__ feats) {
  int t = blockIdx.x * blockDim.x + threadIdx.x;
  if (t >= B_ * N_ * H_) return;
  int h = t & 7; int bn = t >> 3; int n = bn & (N_ - 1); int b = bn >> 9;
  size_t row = (size_t)bn;
  size_t bhn = ((size_t)b * H_ + h) * N_ + n;
  const float* R = rot + row * 9;
  const float* T = trans + row * 3;
  _Float16* f = feats + row * 896;
  const float* prs = rs + bhn * 16;
#pragma unroll
  for (int d = 0; d < 16; ++d) f[h * 16 + d] = (_Float16)prs[d];
  const float* pp = rpts + bhn * 24;
#pragma unroll
  for (int d = 0; d < PV_; ++d) {
    float x = pp[d * 3 + 0] - T[0], y = pp[d * 3 + 1] - T[1], z = pp[d * 3 + 2] - T[2];
    float o0 = x * R[0] + y * R[1] + z * R[2];       // einsum('..c,bnrc->..r')
    float o1 = x * R[3] + y * R[4] + z * R[5];
    float o2 = x * R[6] + y * R[7] + z * R[8];
    f[128 + (h * 8 + d) * 3 + 0] = (_Float16)o0;
    f[128 + (h * 8 + d) * 3 + 1] = (_Float16)o1;
    f[128 + (h * 8 + d) * 3 + 2] = (_Float16)o2;
    f[320 + h * 8 + d] = (_Float16)sqrtf(o0 * o0 + o1 * o1 + o2 * o2 + 1e-8f);
  }
  const float* pr = rpair + bhn * 64;
#pragma unroll
  for (int d = 0; d < 64; ++d) f[384 + h * 64 + d] = (_Float16)pr[d];
}

// =============================================================== host launcher
extern "C" void kernel_launch(void* const* d_in, const int* in_sizes, int n_in,
                              void* d_out, int out_size, void* d_ws, size_t ws_size,
                              hipStream_t stream) {
  (void)in_sizes; (void)n_in; (void)out_size;
  const float* x        = (const float*)d_in[0];
  const float* pairwise = (const float*)d_in[1];
  const float* rot      = (const float*)d_in[2];
  const float* trans    = (const float*)d_in[3];
  const int*   pos      = (const int*)d_in[4];
  const unsigned char* mask = (const unsigned char*)d_in[5];
  const float* Wq_s = (const float*)d_in[6];
  const float* Wk_s = (const float*)d_in[7];
  const float* Wv_s = (const float*)d_in[8];
  const float* Wq_p = (const float*)d_in[9];
  const float* Wk_p = (const float*)d_in[10];
  const float* Wv_p = (const float*)d_in[11];
  const float* point_w = (const float*)d_in[12];
  const float* W_pair  = (const float*)d_in[13];
  const float* b_pair  = (const float*)d_in[14];
  const float* W_out   = (const float*)d_in[15];
  const float* b_out   = (const float*)d_in[16];

  char* base = (char*)d_ws;
  size_t off = 0;
  auto take = [&](size_t bytes) -> void* {
    off = (off + 255) & ~(size_t)255;
    void* p = base + off;
    off += bytes;
    return p;
  };
  const int MR = B_ * N_;                  // 2048 token rows
  const size_t BHN = (size_t)B_ * H_ * N_;
  _Float16* xh    = (_Float16*)take((size_t)MR * DIM_ * 2);
  _Float16* whqs  = (_Float16*)take((size_t)DIM_ * 128 * 2);
  _Float16* whks  = (_Float16*)take((size_t)DIM_ * 128 * 2);
  _Float16* whvs  = (_Float16*)take((size_t)DIM_ * 128 * 2);
  _Float16* whqp  = (_Float16*)take((size_t)DIM_ * 96 * 2);
  _Float16* whkp  = (_Float16*)take((size_t)DIM_ * 96 * 2);
  _Float16* whvp  = (_Float16*)take((size_t)DIM_ * 192 * 2);
  _Float16* whout = (_Float16*)take((size_t)896 * DIM_ * 2);
  float* Pqs = (float*)take((size_t)MR * 128 * 4);
  float* Pks = (float*)take((size_t)MR * 128 * 4);
  float* Pvs = (float*)take((size_t)MR * 128 * 4);
  float* Pqp = (float*)take((size_t)MR * 96 * 4);
  float* Pkp = (float*)take((size_t)MR * 96 * 4);
  float* Pvp = (float*)take((size_t)MR * 192 * 4);
  _Float16* qAb  = (_Float16*)take(BHN * 32 * 2);
  _Float16* kBb  = (_Float16*)take(BHN * 32 * 2);
  _Float16* vshb = (_Float16*)take(BHN * 16 * 2);
  _Float16* vphb = (_Float16*)take(BHN * 24 * 2);
  float* c1b = (float*)take(BHN * 4);
  float* c2b = (float*)take(BHN * 4);
  _Float16* bias8 = (_Float16*)take((size_t)B_ * N_ * N_ * H_ * 2);
  _Float16* attnb = (_Float16*)take(BHN * N_ * 2);
  float* rsb    = (float*)take(BHN * 16 * 4);
  float* rptsb  = (float*)take(BHN * 24 * 4);
  float* rpairb = (float*)take(BHN * 64 * 4);
  _Float16* feats = (_Float16*)take((size_t)MR * 896 * 2);
  if (off > ws_size) return;

  auto cvt = [&](const float* s, _Float16* d, int n) {
    k_f2h<<<(n + 255) / 256, 256, 0, stream>>>(s, d, n);
  };
  cvt(x, xh, MR * DIM_);
  cvt(Wq_s, whqs, DIM_ * 128);
  cvt(Wk_s, whks, DIM_ * 128);
  cvt(Wv_s, whvs, DIM_ * 128);
  cvt(Wq_p, whqp, DIM_ * 96);
  cvt(Wk_p, whkp, DIM_ * 96);
  cvt(Wv_p, whvp, DIM_ * 192);
  cvt(W_out, whout, 896 * DIM_);

  k_gemm_f16<<<(MR / 16) * (128 / 16), 32, 0, stream>>>(xh, whqs, Pqs, nullptr, MR, 128, DIM_);
  k_gemm_f16<<<(MR / 16) * (128 / 16), 32, 0, stream>>>(xh, whks, Pks, nullptr, MR, 128, DIM_);
  k_gemm_f16<<<(MR / 16) * (128 / 16), 32, 0, stream>>>(xh, whvs, Pvs, nullptr, MR, 128, DIM_);
  k_gemm_f16<<<(MR / 16) * (96 / 16),  32, 0, stream>>>(xh, whqp, Pqp, nullptr, MR, 96, DIM_);
  k_gemm_f16<<<(MR / 16) * (96 / 16),  32, 0, stream>>>(xh, whkp, Pkp, nullptr, MR, 96, DIM_);
  k_gemm_f16<<<(MR / 16) * (192 / 16), 32, 0, stream>>>(xh, whvp, Pvp, nullptr, MR, 192, DIM_);

  k_geom<<<(B_ * N_ * H_ + 255) / 256, 256, 0, stream>>>(
      Pqs, Pks, Pvs, Pqp, Pkp, Pvp, rot, trans, pos, point_w,
      qAb, kBb, vshb, vphb, c1b, c2b);

  k_pairbias<<<(B_ * N_ * N_ * H_) / 256, 256, 0, stream>>>(pairwise, W_pair, b_pair, bias8);

  k_attn<<<B_ * H_ * (N_ / 16), 32, 0, stream>>>(
      qAb, kBb, bias8, c1b, c2b, vshb, vphb, mask, attnb, rsb, rptsb);

  k_respair<<<B_ * N_, 32, 0, stream>>>(attnb, pairwise, rpairb);

  k_finalize<<<(B_ * N_ * H_ + 255) / 256, 256, 0, stream>>>(rsb, rptsb, rpairb, rot, trans, feats);

  k_gemm_f16<<<(MR / 16) * (DIM_ / 16), 32, 0, stream>>>(
      feats, whout, (float*)d_out, b_out, MR, DIM_, 896);
}
